// Attention_19327352832430
// MI455X (gfx1250) — matrix operand order repaired and checked by reading, not compiled, not run on hardware
//
#include <hip/hip_runtime.h>

// ---------------- types ----------------
typedef __attribute__((ext_vector_type(8)))  __bf16 v8bf;
typedef __attribute__((ext_vector_type(16))) __bf16 v16bf;
typedef __attribute__((ext_vector_type(8)))  float  v8f;

#define WMMA_BF16(A, B, C) \
  __builtin_amdgcn_wmma_f32_16x16x32_bf16(false, (A), false, (B), (short)0, (C), false, false)

static __device__ __forceinline__ unsigned short f32_to_bf16_bits(float f) {
  unsigned u = __builtin_bit_cast(unsigned, f);
  u += 0x7FFFu + ((u >> 16) & 1u);          // round-to-nearest-even
  return (unsigned short)(u >> 16);
}
static __device__ __forceinline__ v16bf cat8(v8bf lo, v8bf hi) {
  return __builtin_shufflevector(lo, hi, 0,1,2,3,4,5,6,7,8,9,10,11,12,13,14,15);
}
static __device__ __forceinline__ int clampi(int v, int lo, int hi) {
  return v < lo ? lo : (v > hi ? hi : v);
}

// ---------------- problem constants ----------------
constexpr int B_     = 2;
constexpr int N_     = 1024;
constexpr int DIM    = 1024;
constexpr int HEADS  = 16;
constexpr int DHEAD  = 64;
constexpr int INNER  = HEADS * DHEAD;   // 1024
constexpr int MAXPOS = 512;
constexpr int M_     = B_ * N_;         // 2048 rows in projection GEMMs

// ---------------- workspace layout (bytes) ----------------
constexpr size_t XB_OFF   = 0;                                        // x bf16 [2048,1024]
constexpr size_t WQT_OFF  = XB_OFF  + (size_t)M_   * DIM   * 2;       // Wq^T  [1024,1024]
constexpr size_t WKVT_OFF = WQT_OFF + (size_t)INNER* DIM   * 2;       // Wkv^T [2048,1024]
constexpr size_t WOT_OFF  = WKVT_OFF+ (size_t)2*INNER*DIM  * 2;       // Wo^T  [1024,1024]
constexpr size_t REL_OFF  = WOT_OFF + (size_t)DIM  * INNER * 2;       // rel bf16 [1025,64]
constexpr size_t QB_OFF   = (REL_OFF + (size_t)1025*64*2 + 255) & ~(size_t)255;
constexpr size_t KB_OFF   = QB_OFF  + (size_t)M_ * INNER * 2;         // k [b,h,n,d]
constexpr size_t VT_OFF   = KB_OFF  + (size_t)M_ * INNER * 2;         // v^T [b,h,d,n]
constexpr size_t AO_OFF   = VT_OFF  + (size_t)M_ * INNER * 2;         // attn out bf16 [b,n,inner]
// total ~29.5 MB

// ---------------- kernel 1: fp32 -> bf16 (optional transpose) ----------------
__global__ void cvt_bf16_kernel(const float* __restrict__ src, unsigned short* __restrict__ dst,
                                int R, int C, int transpose) {
  size_t total = (size_t)R * C;
  for (size_t i = (size_t)blockIdx.x * blockDim.x + threadIdx.x; i < total;
       i += (size_t)gridDim.x * blockDim.x) {
    int r = (int)(i / C), c = (int)(i % C);
    unsigned short v = f32_to_bf16_bits(src[i]);
    if (transpose) dst[(size_t)c * R + r] = v;
    else           dst[i] = v;
  }
}

// ---------------- kernel 2: fused QKV projection GEMM ----------------
// one 16x64 output block per wave; explicit software pipeline: next-step
// operand loads are issued BEFORE the current step's WMMAs so the matrix
// pipe overlaps load latency (partial loadcnt waits instead of full drains).
__global__ __launch_bounds__(128) void proj_gemm_kernel(
    const unsigned short* __restrict__ xb,
    const unsigned short* __restrict__ wqt,
    const unsigned short* __restrict__ wkvt,
    unsigned short* __restrict__ qb,
    unsigned short* __restrict__ kb,
    unsigned short* __restrict__ vtb) {
  const int lane = threadIdx.x & 31, wid = threadIdx.x >> 5;
  const int half = lane >> 4, lmod = lane & 15;
  const int mt = blockIdx.x;                 // 0..127 : 16-row tile of M=2048
  const int g  = blockIdx.y * 4 + wid;       // 0..47  : 64-col group; 0-15 q, 16-31 k, 32-47 v
  const int cclass = g >> 4;                 // 0=q, 1=k, 2=v
  const int o0 = (g & 15) * 64;              // feature base within class

  const unsigned short* wbase =
      (cclass == 0) ? wqt : (cclass == 1) ? wkvt : (wkvt + (size_t)INNER * DIM);
  const unsigned short* w0 = wbase + (size_t)(o0 +  0 + lmod) * DIM + 16 * half;
  const unsigned short* w1 = wbase + (size_t)(o0 + 16 + lmod) * DIM + 16 * half;
  const unsigned short* w2 = wbase + (size_t)(o0 + 32 + lmod) * DIM + 16 * half;
  const unsigned short* w3 = wbase + (size_t)(o0 + 48 + lmod) * DIM + 16 * half;
  const unsigned short* arow = xb + (size_t)(mt * 16 + lmod) * DIM;

  // prologue: load first operand set
  v8bf  alo = *(const v8bf*)(arow + 16 * half);
  v8bf  ahi = *(const v8bf*)(arow + 8 + 16 * half);
  v16bf b0  = *(const v16bf*)(w0);
  v16bf b1  = *(const v16bf*)(w1);
  v16bf b2  = *(const v16bf*)(w2);
  v16bf b3  = *(const v16bf*)(w3);

  v8f acc0 = {}, acc1 = {}, acc2 = {}, acc3 = {};
  for (int k0 = 0; k0 < DIM - 32; k0 += 32) {
    // rotate current operands into locals
    v16bf a  = cat8(alo, ahi);
    v16bf c0 = b0, c1 = b1, c2 = b2, c3 = b3;
    // issue next-step loads before the WMMAs
    const int kn = k0 + 32;
    alo = *(const v8bf*)(arow + kn + 16 * half);
    ahi = *(const v8bf*)(arow + kn + 8 + 16 * half);
    b0  = *(const v16bf*)(w0 + kn);
    b1  = *(const v16bf*)(w1 + kn);
    b2  = *(const v16bf*)(w2 + kn);
    b3  = *(const v16bf*)(w3 + kn);
    acc0 = WMMA_BF16(a, c0, acc0);
    acc1 = WMMA_BF16(a, c1, acc1);
    acc2 = WMMA_BF16(a, c2, acc2);
    acc3 = WMMA_BF16(a, c3, acc3);
  }
  {
    v16bf a = cat8(alo, ahi);
    acc0 = WMMA_BF16(a, b0, acc0);
    acc1 = WMMA_BF16(a, b1, acc1);
    acc2 = WMMA_BF16(a, b2, acc2);
    acc3 = WMMA_BF16(a, b3, acc3);
  }

  v8f accs[4] = {acc0, acc1, acc2, acc3};
#pragma unroll
  for (int s = 0; s < 4; ++s) {
    const int o = o0 + s * 16 + lmod;        // output feature 0..1023
    const int hh = o >> 6, d = o & 63;
#pragma unroll
    for (int i = 0; i < 8; ++i) {
      int row = mt * 16 + i + 8 * half;      // global m
      int bb = row >> 10, n = row & 1023;
      unsigned short val = f32_to_bf16_bits(accs[s][i]);
      if (cclass == 0)      qb [(((size_t)bb * HEADS + hh) * N_ + n) * DHEAD + d] = val;
      else if (cclass == 1) kb [(((size_t)bb * HEADS + hh) * N_ + n) * DHEAD + d] = val;
      else                  vtb[(((size_t)bb * HEADS + hh) * DHEAD + d) * N_ + n] = val;
    }
  }
}

// ---------------- kernel 3: attention (dots + rel-bias + softmax + P@V) ----------------
constexpr int SSTRIDE = 1032;   // fp32 scores row stride (pad vs LDS bank aliasing)
constexpr int ASTRIDE = 1040;   // bf16 probs row stride
__global__ __launch_bounds__(128) void attn_kernel(
    const unsigned short* __restrict__ qb,
    const unsigned short* __restrict__ kb,
    const unsigned short* __restrict__ vtb,
    const unsigned short* __restrict__ relb,
    unsigned short* __restrict__ aob) {
  __shared__ float scores[16][SSTRIDE];          // 66 KB fp32 scores
  __shared__ unsigned short attnb[16][ASTRIDE];  // 33 KB bf16 probs
  __shared__ float Mbuf[4][16][32];              // per-wave rel-bias tile scratch
  __shared__ float redmax[16][8];
  __shared__ float redsum[16][8];

  const int lane = threadIdx.x & 31, wid = threadIdx.x >> 5;
  const int half = lane >> 4, lmod = lane & 15;
  const int bh = blockIdx.x >> 6;            // b*16+h
  const int n0 = (blockIdx.x & 63) * 16;     // query tile base
  const float scale = 0.125f;                // 64^-0.5

  // Q tile A-operands (K = d = 0..31 and 32..63), loaded once
  const unsigned short* qrow = qb + ((size_t)bh * N_ + n0 + lmod) * DHEAD;
  v16bf a0 = cat8(*(const v8bf*)(qrow + 16 * half),      *(const v8bf*)(qrow + 8 + 16 * half));
  v16bf a1 = cat8(*(const v8bf*)(qrow + 32 + 16 * half), *(const v8bf*)(qrow + 40 + 16 * half));

  // -------- phase 1: scores = (Q.K^T + Q.rel^T) * scale --------
  for (int jt = wid; jt < 64; jt += 4) {
    const unsigned short* krow = kb + ((size_t)bh * N_ + jt * 16 + lmod) * DHEAD;
    const int base = n0 - jt * 16 - 15 + MAXPOS;   // diag offset for t=0
    const unsigned short* rr0 = relb + (size_t)clampi(base + lmod, 0, 2 * MAXPOS) * DHEAD;
    const unsigned short* rr1 = relb + (size_t)clampi(base + 16 + lmod, 0, 2 * MAXPOS) * DHEAD;

    // issue all 6 B-operand loads up front
    v16bf kb0 = *(const v16bf*)(krow + 16 * half);
    v16bf kb1 = *(const v16bf*)(krow + 32 + 16 * half);
    v16bf rb00 = *(const v16bf*)(rr0 + 16 * half);
    v16bf rb01 = *(const v16bf*)(rr0 + 32 + 16 * half);
    v16bf rb10 = *(const v16bf*)(rr1 + 16 * half);
    v16bf rb11 = *(const v16bf*)(rr1 + 32 + 16 * half);

    v8f c = {}, m0 = {}, m1 = {};
    c  = WMMA_BF16(a0, kb0,  c);
    c  = WMMA_BF16(a1, kb1,  c);
    m0 = WMMA_BF16(a0, rb00, m0);
    m0 = WMMA_BF16(a1, rb01, m0);
    m1 = WMMA_BF16(a0, rb10, m1);
    m1 = WMMA_BF16(a1, rb11, m1);

#pragma unroll
    for (int i = 0; i < 8; ++i) {
      int row = i + 8 * half;
      Mbuf[wid][row][lmod]      = m0[i];
      Mbuf[wid][row][16 + lmod] = m1[i];
    }
    // diagonal gather: pos[i][j] = M[i][i-j+15]  (DS ops in-order within wave)
#pragma unroll
    for (int i = 0; i < 8; ++i) {
      int row = i + 8 * half;
      int t = row - lmod + 15;                       // in [0,30]
      scores[row][jt * 16 + lmod] = (c[i] + Mbuf[wid][row][t]) * scale;
    }
  }
  __syncthreads();

  // -------- phase 2: row softmax over 1024 cols (8 threads per row),
  //          final pass writes normalized probs as bf16 --------
  {
    const int srow = threadIdx.x >> 3, seg = threadIdx.x & 7, cb = seg * 128;
    float lmax = -1e30f;
    for (int c = 0; c < 128; ++c) lmax = fmaxf(lmax, scores[srow][cb + c]);
    redmax[srow][seg] = lmax;
    __syncthreads();
    float rmax = redmax[srow][0];
#pragma unroll
    for (int s = 1; s < 8; ++s) rmax = fmaxf(rmax, redmax[srow][s]);
    float lsum = 0.f;
    for (int c = 0; c < 128; ++c) {
      float e = __expf(scores[srow][cb + c] - rmax);
      scores[srow][cb + c] = e;
      lsum += e;
    }
    redsum[srow][seg] = lsum;
    __syncthreads();
    float rsum = redsum[srow][0];
#pragma unroll
    for (int s = 1; s < 8; ++s) rsum += redsum[srow][s];
    float inv = 1.f / rsum;
    for (int c = 0; c < 128; ++c)
      attnb[srow][cb + c] = f32_to_bf16_bits(scores[srow][cb + c] * inv);
  }
  __syncthreads();

  // -------- phase 3: O = P @ V  (wave wid handles d-block wid*16..+15),
  //          software-pipelined like the GEMMs --------
  const unsigned short* vbase =
      vtb + ((size_t)bh * DHEAD + wid * 16 + lmod) * N_ + 16 * half;
  v8bf  plo = *(const v8bf*)(&attnb[lmod][0] + 16 * half);
  v8bf  phi = *(const v8bf*)(&attnb[lmod][0] + 8 + 16 * half);
  v16bf bm  = *(const v16bf*)(vbase);
  v8f o = {};
  for (int kk = 0; kk < N_ - 32; kk += 32) {
    v16bf a  = cat8(plo, phi);
    v16bf bc = bm;
    const int kn = kk + 32;
    plo = *(const v8bf*)(&attnb[lmod][kn] + 16 * half);
    phi = *(const v8bf*)(&attnb[lmod][kn] + 8 + 16 * half);
    bm  = *(const v16bf*)(vbase + kn);
    o = WMMA_BF16(a, bc, o);
  }
  o = WMMA_BF16(cat8(plo, phi), bm, o);

  const int b = bh >> 4, h = bh & 15;
#pragma unroll
  for (int i = 0; i < 8; ++i) {
    int row = i + 8 * half;
    aob[((size_t)b * N_ + n0 + row) * INNER + h * DHEAD + wid * 16 + lmod] =
        f32_to_bf16_bits(o[i]);
  }
}

// ---------------- kernel 4: output projection + bias ----------------
// one 16x64 output block per wave; same explicit pipeline as proj GEMM.
__global__ __launch_bounds__(128) void out_gemm_kernel(
    const unsigned short* __restrict__ aob,
    const unsigned short* __restrict__ wot,
    const float* __restrict__ bo,
    float* __restrict__ out) {
  const int lane = threadIdx.x & 31, wid = threadIdx.x >> 5;
  const int half = lane >> 4, lmod = lane & 15;
  const int mt = blockIdx.x;                  // 0..127
  const int g  = blockIdx.y * 4 + wid;        // 0..15 : 64-col group
  const unsigned short* arow = aob + (size_t)(mt * 16 + lmod) * INNER;
  const unsigned short* w0 = wot + (size_t)(g * 64 +  0 + lmod) * INNER + 16 * half;
  const unsigned short* w1 = wot + (size_t)(g * 64 + 16 + lmod) * INNER + 16 * half;
  const unsigned short* w2 = wot + (size_t)(g * 64 + 32 + lmod) * INNER + 16 * half;
  const unsigned short* w3 = wot + (size_t)(g * 64 + 48 + lmod) * INNER + 16 * half;

  v8bf  alo = *(const v8bf*)(arow + 16 * half);
  v8bf  ahi = *(const v8bf*)(arow + 8 + 16 * half);
  v16bf b0  = *(const v16bf*)(w0);
  v16bf b1  = *(const v16bf*)(w1);
  v16bf b2  = *(const v16bf*)(w2);
  v16bf b3  = *(const v16bf*)(w3);

  v8f acc0 = {}, acc1 = {}, acc2 = {}, acc3 = {};
  for (int k0 = 0; k0 < INNER - 32; k0 += 32) {
    v16bf a  = cat8(alo, ahi);
    v16bf c0 = b0, c1 = b1, c2 = b2, c3 = b3;
    const int kn = k0 + 32;
    alo = *(const v8bf*)(arow + kn + 16 * half);
    ahi = *(const v8bf*)(arow + kn + 8 + 16 * half);
    b0  = *(const v16bf*)(w0 + kn);
    b1  = *(const v16bf*)(w1 + kn);
    b2  = *(const v16bf*)(w2 + kn);
    b3  = *(const v16bf*)(w3 + kn);
    acc0 = WMMA_BF16(a, c0, acc0);
    acc1 = WMMA_BF16(a, c1, acc1);
    acc2 = WMMA_BF16(a, c2, acc2);
    acc3 = WMMA_BF16(a, c3, acc3);
  }
  {
    v16bf a = cat8(alo, ahi);
    acc0 = WMMA_BF16(a, b0, acc0);
    acc1 = WMMA_BF16(a, b1, acc1);
    acc2 = WMMA_BF16(a, b2, acc2);
    acc3 = WMMA_BF16(a, b3, acc3);
  }

  v8f accs[4] = {acc0, acc1, acc2, acc3};
#pragma unroll
  for (int s = 0; s < 4; ++s) {
    const int col = g * 64 + s * 16 + lmod;
    const float bias = bo[col];
#pragma unroll
    for (int i = 0; i < 8; ++i) {
      int row = mt * 16 + i + 8 * half;
      out[(size_t)row * DIM + col] = accs[s][i] + bias;
    }
  }
}

// ---------------- host ----------------
extern "C" void kernel_launch(void* const* d_in, const int* in_sizes, int n_in,
                              void* d_out, int out_size, void* d_ws, size_t ws_size,
                              hipStream_t stream) {
  const float* x    = (const float*)d_in[0];
  const float* Wq   = (const float*)d_in[1];
  const float* Wkv  = (const float*)d_in[2];
  const float* Wo   = (const float*)d_in[3];
  const float* bo   = (const float*)d_in[4];
  const float* rel  = (const float*)d_in[5];

  char* ws = (char*)d_ws;
  unsigned short* xb   = (unsigned short*)(ws + XB_OFF);
  unsigned short* wqt  = (unsigned short*)(ws + WQT_OFF);
  unsigned short* wkvt = (unsigned short*)(ws + WKVT_OFF);
  unsigned short* wot  = (unsigned short*)(ws + WOT_OFF);
  unsigned short* relb = (unsigned short*)(ws + REL_OFF);
  unsigned short* qb   = (unsigned short*)(ws + QB_OFF);
  unsigned short* kb   = (unsigned short*)(ws + KB_OFF);
  unsigned short* vtb  = (unsigned short*)(ws + VT_OFF);
  unsigned short* aob  = (unsigned short*)(ws + AO_OFF);

  // bf16 conversion (+ weight transposes so WMMA B-operands are contiguous)
  cvt_bf16_kernel<<<2048, 256, 0, stream>>>(x,   xb,   M_,   DIM,      0);
  cvt_bf16_kernel<<<1024, 256, 0, stream>>>(Wq,  wqt,  DIM,  INNER,    1);
  cvt_bf16_kernel<<<2048, 256, 0, stream>>>(Wkv, wkvt, DIM,  2*INNER,  1);
  cvt_bf16_kernel<<<1024, 256, 0, stream>>>(Wo,  wot,  INNER, DIM,     1);
  cvt_bf16_kernel<<<128,  256, 0, stream>>>(rel, relb, 2*MAXPOS+1, DHEAD, 0);

  // fused QKV projection: 128 M-tiles x 48 col-groups (16x64 per wave)
  proj_gemm_kernel<<<dim3(128, 12), 128, 0, stream>>>(xb, wqt, wkvt, qb, kb, vtb);

  // attention: one block per (b,h,16-query-row tile)
  attn_kernel<<<B_ * HEADS * (N_ / 16), 128, 0, stream>>>(qb, kb, vtb, relb, aob);

  // output projection + bias -> fp32 (16x64 per wave)
  out_gemm_kernel<<<dim3(128, 4), 128, 0, stream>>>(aob, wot, bo, (float*)d_out);
}
